// Relu2Attention_84868553769164
// MI455X (gfx1250) — compile-verified, matrix operand order checked by reading, not run on hardware
//
#include <hip/hip_runtime.h>
#include <hip/hip_bf16.h>
#include <cstdint>

// ---------------- problem constants ----------------
#define TT      4096
#define NE      2048
#define QDIM    2048
#define DKV     1024
#define KDIM    512
#define NH      16
#define KVH     4
#define DH      128
#define DSTATE  16
#define DTR     64
#define XDBL    96          // DTR + 2*DSTATE
#define WIN     512
#define OUTDIM  (QDIM + DKV)   // 3072

typedef __attribute__((ext_vector_type(8)))  float  v8f;
typedef __attribute__((ext_vector_type(16))) __bf16 v16bf;

// ---------------- bf16 helpers ----------------
__device__ __forceinline__ unsigned short f2bf(float f) {
    union { float f; unsigned u; } v; v.f = f;
    unsigned r = v.u + 0x7FFFu + ((v.u >> 16) & 1u);   // RNE
    return (unsigned short)(r >> 16);
}

__device__ __forceinline__ v8f vzero8() {
    v8f z; for (int i = 0; i < 8; ++i) z[i] = 0.0f; return z;
}

// ---------------- WMMA fragment loaders ----------------
// A-frag (16x32 bf16, row-major src) and B-frag from an NT operand
// (C[m,n] = sum_k A[m,k] * W[n,k]) share the identical per-lane pattern:
//   lanes 0-15:  row r0+lane,    K = {k0..k0+7,  k0+16..k0+23}
//   lanes 16-31: row r0+lane-16, K = {k0+8..k0+15, k0+24..k0+31}
__device__ __forceinline__ v16bf ld_frag_nt(const unsigned short* __restrict__ base,
                                            int ld, int r0, int k0, int lane, int rmax) {
    int r  = r0 + (lane & 15); if (r > rmax) r = rmax;
    int kb = k0 + ((lane & 16) ? 8 : 0);
    const unsigned short* p = base + (size_t)r * ld + kb;
    union { v16bf v; uint4 q[2]; } u;
    u.q[0] = *(const uint4*)(p);
    u.q[1] = *(const uint4*)(p + 16);
    return u.v;
}

// B-frag where the WMMA-K dim runs along LDS rows (P*V: B[k=key,n=hd] = Vt[key,hd]).
// Vt is a 32x128 bf16 row-major tile resident in LDS.
__device__ __forceinline__ v16bf ld_frag_kmajor_lds(const unsigned short* Vt,
                                                    int n0, int lane) {
    int col = n0 + (lane & 15);
    int kb  = (lane & 16) ? 8 : 0;
    union { v16bf v; unsigned short s[16]; } u;
#pragma unroll
    for (int i = 0; i < 8; ++i)      u.s[i]     = Vt[(kb + i) * DH + col];
#pragma unroll
    for (int i = 0; i < 8; ++i)      u.s[8 + i] = Vt[(kb + 16 + i) * DH + col];
    return u.v;
}

// ---------------- kernel: fp32 -> bf16 convert ----------------
__global__ void k_cvt_bf16(const float* __restrict__ in, unsigned short* __restrict__ out, int n) {
    for (int i = blockIdx.x * blockDim.x + threadIdx.x; i < n; i += gridDim.x * blockDim.x)
        out[i] = f2bf(in[i]);
}

// ---------------- kernel: bf16 WMMA GEMM, C = A * B^T ----------------
// A: MxK row-major bf16 (lda), Bw: NxK row-major bf16 (ldb)
// Wave tile 32x64 (2 A-frags x 4 B-frags, 8 accumulators); block = 8 waves (2Mx4N)
// -> block tile 64x256. Columns n < split -> bf16 outBf; n >= split -> f32 outF.
__global__ void __launch_bounds__(256)
k_gemm_nt(const unsigned short* __restrict__ A, const unsigned short* __restrict__ Bw,
          int K, int lda, int ldb,
          unsigned short* __restrict__ outBf, int ldbf,
          float* __restrict__ outF, int ldf, int split) {
    const int lane = threadIdx.x & 31;
    const int wave = threadIdx.x >> 5;          // 0..7, arranged 2(M) x 4(N)
    const int m0 = blockIdx.x * 64  + (wave >> 2) * 32;
    const int n0 = blockIdx.y * 256 + (wave & 3) * 64;

    v8f acc[2][4];
#pragma unroll
    for (int i = 0; i < 2; ++i)
#pragma unroll
        for (int j = 0; j < 4; ++j) acc[i][j] = vzero8();

    for (int k = 0; k < K; k += 32) {
        if (k + 64 < K) {
            __builtin_prefetch(A  + (size_t)(m0 + (lane & 15)) * lda + k + 64, 0, 1);
            __builtin_prefetch(Bw + (size_t)(n0 + (lane & 15)) * ldb + k + 64, 0, 1);
        }
        v16bf a0 = ld_frag_nt(A, lda, m0,      k, lane, 1 << 30);
        v16bf a1 = ld_frag_nt(A, lda, m0 + 16, k, lane, 1 << 30);
        v16bf b[4];
#pragma unroll
        for (int j = 0; j < 4; ++j)
            b[j] = ld_frag_nt(Bw, ldb, n0 + j * 16, k, lane, 1 << 30);
#pragma unroll
        for (int j = 0; j < 4; ++j) {
            acc[0][j] = __builtin_amdgcn_wmma_f32_16x16x32_bf16(false, a0, false, b[j], (short)0, acc[0][j], false, false);
            acc[1][j] = __builtin_amdgcn_wmma_f32_16x16x32_bf16(false, a1, false, b[j], (short)0, acc[1][j], false, false);
        }
    }

    const int half = (lane & 16) ? 8 : 0;
    const int nl   = lane & 15;
#pragma unroll
    for (int i = 0; i < 2; ++i) {
#pragma unroll
        for (int j = 0; j < 4; ++j) {
            v8f c = acc[i][j];
            int ncol = n0 + j * 16 + nl;
#pragma unroll
            for (int v = 0; v < 8; ++v) {
                int m = m0 + i * 16 + half + v;
                if (ncol < split) outBf[(size_t)m * ldbf + ncol] = f2bf(c[v]);
                else              outF [(size_t)m * ldf  + (ncol - split)] = c[v];
            }
        }
    }
}

// ---------------- kernel: x_dbl = kv @ x_proj_w^T  (T x 96) ----------------
__global__ void k_xproj(const float* __restrict__ kv, const float* __restrict__ xw,
                        float* __restrict__ xd) {
    int idx = blockIdx.x * blockDim.x + threadIdx.x;
    if (idx >= TT * XDBL) return;
    int t = idx / XDBL, e = idx % XDBL;
    const float4* u = (const float4*)(kv + (size_t)t * DKV);
    const float4* w = (const float4*)(xw + (size_t)e * DKV);
    float s = 0.f;
    for (int d = 0; d < DKV / 4; ++d) {
        float4 a = u[d], b = w[d];
        s += a.x * b.x + a.y * b.y + a.z * b.z + a.w * b.w;
    }
    xd[(size_t)t * XDBL + e] = s;
}

// ---------------- kernel: delta = softplus(dt @ dt_proj_w^T + b) ----------------
__global__ void k_delta(const float* __restrict__ xd, const float* __restrict__ dtw,
                        const float* __restrict__ dtb, float* __restrict__ delta) {
    int idx = blockIdx.x * blockDim.x + threadIdx.x;
    if (idx >= TT * DKV) return;
    int t = idx >> 10, d = idx & (DKV - 1);
    const float4* a = (const float4*)(xd + (size_t)t * XDBL);       // first 64 cols = dt
    const float4* w = (const float4*)(dtw + (size_t)d * DTR);
    float s = dtb[d];
    for (int r = 0; r < DTR / 4; ++r) {
        float4 x = a[r], y = w[r];
        s += x.x * y.x + x.y * y.y + x.z * y.z + x.w * y.w;
    }
    float sp = (s > 20.f) ? s : log1pf(__expf(s));
    delta[(size_t)t * DKV + d] = sp;
}

// ---------------- kernel: S6 selective scan (sequential over T) ----------------
// 16 lanes per d-channel (n = 0..15), 16 channels per 256-thread block.
__global__ void __launch_bounds__(256)
k_scan(const float* __restrict__ kv, const float* __restrict__ delta,
       const float* __restrict__ xd, const float* __restrict__ A_log,
       const float* __restrict__ Dp, unsigned short* __restrict__ kvs) {
    const int tid = threadIdx.x;
    const int grp = tid >> 4;          // channel slot in block
    const int n   = tid & 15;          // state index
    const int d   = blockIdx.x * 16 + grp;

    const float A  = -__expf(A_log[(size_t)d * DSTATE + n]);
    const float Dd = Dp[d];
    float h = 0.f;

    const float* pdel = delta + d;
    const float* pu   = kv + d;
    const float* pB   = xd + DTR + n;
    const float* pC   = xd + DTR + DSTATE + n;
    unsigned short* py = kvs + d;

    for (int t = 0; t < TT; ++t) {
        float dt = pdel[(size_t)t * DKV];
        float u  = pu  [(size_t)t * DKV];
        float Bn = pB  [(size_t)t * XDBL];
        float Cn = pC  [(size_t)t * XDBL];
        h = __expf(dt * A) * h + dt * Bn * u;
        float p = h * Cn;
        p += __shfl_xor(p, 1, 32);
        p += __shfl_xor(p, 2, 32);
        p += __shfl_xor(p, 4, 32);
        p += __shfl_xor(p, 8, 32);
        if (n == 0) py[(size_t)t * DKV] = f2bf(p + Dd * u);
    }
}

// ---------------- kernel: flash sliding-window attention (bf16 WMMA) ----------------
// grid: (T/64, NH); block 128 = 4 waves; each wave owns a 16-row query tile.
// V tiles are staged per-wave into LDS with CDNA5 async-to-LDS copies.
__global__ void __launch_bounds__(128)
k_flash(const unsigned short* __restrict__ Qb, const unsigned short* __restrict__ KVs,
        unsigned short* __restrict__ Yb) {
    __shared__ unsigned short Pl[4][16 * 32];       // P tiles, 1KB per wave
    __shared__ unsigned short Vt[4][32 * DH];       // V tiles, 8KB per wave

    const int lane = threadIdx.x & 31;
    const int wave = threadIdx.x >> 5;              // 0..3
    const int head = blockIdx.y;
    const int kvh  = head >> 2;
    const int i0   = blockIdx.x * 64 + wave * 16;

    const unsigned short* Qh = Qb  + head * DH;                 // ld = QDIM
    const unsigned short* Kh = KVs + kvh * DH;                  // ld = DKV
    const unsigned short* Vh = KVs + KDIM + kvh * DH;           // ld = DKV
    unsigned short* Pw = &Pl[wave][0];
    unsigned short* Vw = &Vt[wave][0];
    const unsigned vw_lds = (unsigned)(uintptr_t)Vw;            // LDS byte offset

    const float SCALE = 0.08838834764831845f;                   // 1/sqrt(128)

    // Q tile: 4 A-fragments over head dim
    v16bf qa[4];
#pragma unroll
    for (int kk = 0; kk < 4; ++kk)
        qa[kk] = ld_frag_nt(Qh, QDIM, i0, kk * 32, lane, TT - 1);

    v8f O[8]; for (int f = 0; f < 8; ++f) O[f] = vzero8();
    v8f m8, l8;
    for (int v = 0; v < 8; ++v) { m8[v] = -1e30f; l8[v] = 0.f; }

    const int half = (lane & 16) ? 8 : 0;
    const int nl   = lane & 15;

    int jstart = i0 - (WIN - 1); if (jstart < 0) jstart = 0;
    jstart &= ~31;
    const int jend = i0 + 15;

    for (int j = jstart; j <= jend; j += 32) {
        // ---- async-stage the 32x128 bf16 V tile into this wave's LDS region ----
        // 8KB tile = 512 x 16B chunks; 16 chunks per lane, coalesced along hd.
#pragma unroll
        for (int c = 0; c < 16; ++c) {
            int idx  = c * 32 + lane;
            int row  = idx >> 4;                    // key row in tile (16 chunks/row)
            int col0 = (idx & 15) * 8;              // bf16 element col
            int kr   = j + row; if (kr > TT - 1) kr = TT - 1;
            unsigned long long ga =
                (unsigned long long)(uintptr_t)(Vh + (size_t)kr * DKV + col0);
            unsigned la = vw_lds + (unsigned)(row * DH + col0) * 2u;
            asm volatile("global_load_async_to_lds_b128 %0, %1, off"
                         :: "v"(la), "v"(ga) : "memory");
        }

        // ---- scores: two 16-key tiles (overlaps with async V staging) ----
        v8f s0 = vzero8(), s1 = vzero8();
#pragma unroll
        for (int kk = 0; kk < 4; ++kk) {
            v16bf b0 = ld_frag_nt(Kh, DKV, j,      kk * 32, lane, TT - 1);
            v16bf b1 = ld_frag_nt(Kh, DKV, j + 16, kk * 32, lane, TT - 1);
            s0 = __builtin_amdgcn_wmma_f32_16x16x32_bf16(false, qa[kk], false, b0, (short)0, s0, false, false);
            s1 = __builtin_amdgcn_wmma_f32_16x16x32_bf16(false, qa[kk], false, b1, (short)0, s1, false, false);
        }

        // ---- scale + sliding-window causal mask ----
#pragma unroll
        for (int v = 0; v < 8; ++v) {
            int row = i0 + half + v;
            int c0  = j + nl, c1 = c0 + 16;
            bool ok0 = (c0 <= row) && (row - c0 < WIN);
            bool ok1 = (c1 <= row) && (row - c1 < WIN);
            s0[v] = ok0 ? s0[v] * SCALE : -1e30f;
            s1[v] = ok1 ? s1[v] * SCALE : -1e30f;
        }

        // ---- online softmax ----
        v8f mx;
#pragma unroll
        for (int v = 0; v < 8; ++v) mx[v] = fmaxf(s0[v], s1[v]);
#pragma unroll
        for (int msk = 1; msk < 16; msk <<= 1)
#pragma unroll
            for (int v = 0; v < 8; ++v) mx[v] = fmaxf(mx[v], __shfl_xor(mx[v], msk, 32));

        v8f mnew, alpha, p0, p1, rs;
#pragma unroll
        for (int v = 0; v < 8; ++v) {
            mnew[v]  = fmaxf(m8[v], mx[v]);
            alpha[v] = __expf(m8[v] - mnew[v]);
            p0[v]    = __expf(s0[v] - mnew[v]);
            p1[v]    = __expf(s1[v] - mnew[v]);
            rs[v]    = p0[v] + p1[v];
        }
#pragma unroll
        for (int msk = 1; msk < 16; msk <<= 1)
#pragma unroll
            for (int v = 0; v < 8; ++v) rs[v] += __shfl_xor(rs[v], msk, 32);
#pragma unroll
        for (int v = 0; v < 8; ++v) { l8[v] = l8[v] * alpha[v] + rs[v]; m8[v] = mnew[v]; }
#pragma unroll
        for (int f = 0; f < 8; ++f)
#pragma unroll
            for (int v = 0; v < 8; ++v) O[f][v] *= alpha[v];

        // ---- P (D-layout) -> LDS -> A-layout fragment ----
#pragma unroll
        for (int v = 0; v < 8; ++v) {
            int rl = half + v;
            Pw[rl * 32 + nl]      = f2bf(p0[v]);
            Pw[rl * 32 + 16 + nl] = f2bf(p1[v]);
        }
        asm volatile("s_wait_dscnt 0" ::: "memory");
        v16bf pa = ld_frag_nt(Pw, 32, 0, 0, lane, 15);

        // ---- wait for V tile, then O += P * V from LDS ----
        asm volatile("s_wait_asynccnt 0" ::: "memory");
#pragma unroll
        for (int f = 0; f < 8; ++f) {
            v16bf bv = ld_frag_kmajor_lds(Vw, f * 16, lane);
            O[f] = __builtin_amdgcn_wmma_f32_16x16x32_bf16(false, pa, false, bv, (short)0, O[f], false, false);
        }
    }

    // ---- epilogue: normalize + store bf16 ----
    v8f inv;
#pragma unroll
    for (int v = 0; v < 8; ++v) inv[v] = 1.0f / l8[v];
#pragma unroll
    for (int f = 0; f < 8; ++f)
#pragma unroll
        for (int v = 0; v < 8; ++v) {
            int row = i0 + half + v;
            Yb[(size_t)row * QDIM + head * DH + f * 16 + nl] = f2bf(O[f][v] * inv[v]);
        }
}

// ---------------- host launch ----------------
extern "C" void kernel_launch(void* const* d_in, const int* in_sizes, int n_in,
                              void* d_out, int out_size, void* d_ws, size_t ws_size,
                              hipStream_t stream) {
    (void)in_sizes; (void)n_in; (void)out_size; (void)ws_size;
    const float* x    = (const float*)d_in[0];
    const float* Wat  = (const float*)d_in[1];
    const float* Alog = (const float*)d_in[2];
    const float* xw   = (const float*)d_in[3];
    const float* dtw  = (const float*)d_in[4];
    const float* dtb  = (const float*)d_in[5];
    const float* Dp   = (const float*)d_in[6];
    const float* Wo   = (const float*)d_in[7];

    char* ws = (char*)d_ws;
    size_t off = 0;
    auto carve = [&](size_t bytes) { char* p = ws + off; off += (bytes + 255) & ~(size_t)255; return p; };

    unsigned short* Xb  = (unsigned short*)carve((size_t)TT * NE * 2);       // reused as YA
    unsigned short* Wab = (unsigned short*)carve((size_t)OUTDIM * NE * 2);
    unsigned short* Wob = (unsigned short*)carve((size_t)NE * QDIM * 2);
    unsigned short* Qb  = (unsigned short*)carve((size_t)TT * QDIM * 2);
    float*          KV  = (float*)         carve((size_t)TT * DKV * 4);
    float*          XD  = (float*)         carve((size_t)TT * XDBL * 4);
    float*          DL  = (float*)         carve((size_t)TT * DKV * 4);
    unsigned short* KVS = (unsigned short*)carve((size_t)TT * DKV * 2);
    unsigned short* YA  = Xb;   // x is dead after GEMM1

    // 1) bf16 conversions
    k_cvt_bf16<<<4096, 256, 0, stream>>>(x,   Xb,  TT * NE);
    k_cvt_bf16<<<4096, 256, 0, stream>>>(Wat, Wab, OUTDIM * NE);
    k_cvt_bf16<<<4096, 256, 0, stream>>>(Wo,  Wob, NE * QDIM);

    // 2) QKV projection: q -> bf16 (cols < 2048), kv -> f32
    {
        dim3 g(TT / 64, OUTDIM / 256);
        k_gemm_nt<<<g, 256, 0, stream>>>(Xb, Wab, NE, NE, NE, Qb, QDIM, KV, DKV, QDIM);
    }

    // 3) S6 scan pipeline
    k_xproj<<<(TT * XDBL + 255) / 256, 256, 0, stream>>>(KV, xw, XD);
    k_delta<<<(TT * DKV + 255) / 256, 256, 0, stream>>>(XD, dtw, dtb, DL);
    k_scan <<<DKV / 16, 256, 0, stream>>>(KV, DL, XD, Alog, Dp, KVS);

    // 4) sliding-window flash attention
    {
        dim3 g(TT / 64, NH);
        k_flash<<<g, 128, 0, stream>>>(Qb, KVS, YA);
    }

    // 5) output projection -> f32 d_out
    {
        dim3 g(TT / 64, NE / 256);
        k_gemm_nt<<<g, 256, 0, stream>>>(YA, Wob, QDIM, QDIM, NE,
                                         nullptr, 0, (float*)d_out, NE, 0);
    }
}
